// TransformerBlock_26474178412581
// MI455X (gfx1250) — matrix operand order repaired and checked by reading, not compiled, not run on hardware
//
#include <hip/hip_runtime.h>
#include <hip/hip_bf16.h>
#include <math.h>

typedef __attribute__((ext_vector_type(16))) __bf16 bf16x16;
typedef __attribute__((ext_vector_type(8)))  __bf16 bf16x8;
typedef __attribute__((ext_vector_type(8)))  float  v8f;

#define NH 16
#define HD 64

// ---------------------------------------------------------------------------
// TDM support (guarded: 5-arg builtin on ROCm7.2/clang-22, 6-arg on clang-23+)
// ---------------------------------------------------------------------------
#if defined(__has_builtin)
#if __has_builtin(__builtin_amdgcn_tensor_load_to_lds) && __has_builtin(__builtin_amdgcn_s_wait_tensorcnt)
#define USE_TDM 1
#endif
#endif
#ifndef USE_TDM
#define USE_TDM 0
#endif

#if USE_TDM
typedef __attribute__((ext_vector_type(4))) unsigned int u32x4;
typedef __attribute__((ext_vector_type(8))) int i32x8;
typedef __attribute__((ext_vector_type(4))) int i32x4;

// 2-D tile load: tile_w (elems, 2B each) x tile_h rows, row stride in elems.
// LDS pad: after every 16 DWORDs (one 32-elem row) insert 4 DWORDs -> 80B rows.
__device__ __forceinline__ void tdm_load_tile_2d(unsigned lds_off, const void* gptr,
                                                 unsigned tile_w, unsigned tile_h,
                                                 unsigned long long row_stride) {
    unsigned long long ga = (unsigned long long)(uintptr_t)gptr;
    u32x4 g0;
    g0[0] = 1u;                                                   // count=1 (valid user D#)
    g0[1] = lds_off;                                              // lds_addr
    g0[2] = (unsigned)(ga & 0xFFFFFFFFu);                         // global_addr lo
    g0[3] = (unsigned)((ga >> 32) & 0x01FFFFFFu) | (2u << 30);    // global_addr hi | type=2
    const unsigned dim0 = 0x40000000u, dim1 = 0x40000000u;        // huge: no OOB clipping
    i32x8 g1;
    g1[0] = (int)((1u << 16) | (1u << 20) | (3u << 22) | (3u << 25)); // 2B elems, pad en, 16DW/+4DW
    g1[1] = (int)((dim0 & 0xFFFFu) << 16);
    g1[2] = (int)(((dim0 >> 16) & 0xFFFFu) | ((dim1 & 0xFFFFu) << 16));
    g1[3] = (int)(((dim1 >> 16) & 0xFFFFu) | (tile_w << 16));     // tile_dim0
    g1[4] = (int)(tile_h & 0xFFFFu);                              // tile_dim1 (tile_dim2=0)
    g1[5] = (int)(unsigned)(row_stride & 0xFFFFFFFFull);          // tensor_dim0_stride lo
    g1[6] = (int)(unsigned)((row_stride >> 32) & 0xFFFFull);      // stride hi (dim1_stride=0)
    g1[7] = 0;
    i32x4 z4 = {0, 0, 0, 0};
#if __clang_major__ >= 23
    i32x8 z8 = {0, 0, 0, 0, 0, 0, 0, 0};
    __builtin_amdgcn_tensor_load_to_lds(g0, g1, z4, z4, z8, 0);
#else
    __builtin_amdgcn_tensor_load_to_lds(g0, g1, z4, z4, 0);
#endif
}
#endif

// ---------------------------------------------------------------------------
// Kernel 1: ss = silu(t_vec) @ Wss + bss  for both Wss1 and Wss2 (tiny).
// ---------------------------------------------------------------------------
__global__ void cond_kernel(const float* __restrict__ t,
                            const float* __restrict__ Wss1, const float* __restrict__ bss1,
                            const float* __restrict__ Wss2, const float* __restrict__ bss2,
                            float* __restrict__ ss1, float* __restrict__ ss2,
                            int B, int H) {
    int idx = blockIdx.x * blockDim.x + threadIdx.x;
    if (idx >= B * 2 * H) return;
    int b = idx / (2 * H);
    int j = idx % (2 * H);
    float a1 = 0.f, a2 = 0.f;
    for (int i = 0; i < H; ++i) {
        float tv = t[b * H + i];
        float sv = tv / (1.f + __expf(-tv));
        a1 += sv * Wss1[(size_t)i * (2 * H) + j];
        a2 += sv * Wss2[(size_t)i * (2 * H) + j];
    }
    ss1[idx] = a1 + bss1[j];
    ss2[idx] = a2 + bss2[j];
}

// ---------------------------------------------------------------------------
// Kernel 2: comb[b,q,k] = mask[q,k]+bias[q,k] (+kpm penalty), bf16, L2-resident.
// ---------------------------------------------------------------------------
__global__ __launch_bounds__(256) void biascomb_kernel(const float* __restrict__ amask,
                                                       const float* __restrict__ abias,
                                                       const unsigned char* __restrict__ kpm,
                                                       __bf16* __restrict__ comb, int B, int L) {
    size_t idx = (size_t)blockIdx.x * 256 + threadIdx.x;     // over B*L*L
    int k = (int)(idx % L);
    size_t rest = idx / L;
    int q = (int)(rest % L);
    int b = (int)(rest / L);
    float v = amask[(size_t)q * L + k] + abias[(size_t)q * L + k];
    if (kpm[b * L + k]) v = -3.0e38f;
    comb[idx] = (__bf16)v;
}

// ---------------------------------------------------------------------------
// Kernel 3: adaLN: h = (1+s)*LN(x)+sh -> bf16. One block per row.
// ---------------------------------------------------------------------------
__global__ __launch_bounds__(256) void adaln_kernel(const float* __restrict__ x,
                                                    const float* __restrict__ ss,
                                                    __bf16* __restrict__ outh,
                                                    int L, int H) {
    int row = blockIdx.x;
    int b   = row / L;
    int tid = threadIdx.x;
    __shared__ float red[16];

    float v[4], s1 = 0.f, s2 = 0.f;
    for (int i = 0; i < 4; ++i) {
        v[i] = x[(size_t)row * H + tid * 4 + i];
        s1 += v[i];
        s2 += v[i] * v[i];
    }
    for (int off = 1; off < 32; off <<= 1) {
        s1 += __shfl_xor(s1, off, 32);
        s2 += __shfl_xor(s2, off, 32);
    }
    int wave = tid >> 5, lane = tid & 31;
    if (lane == 0) { red[wave] = s1; red[8 + wave] = s2; }
    __syncthreads();
    if (wave == 0) {
        float a  = (lane < 8) ? red[lane]     : 0.f;
        float bb = (lane < 8) ? red[8 + lane] : 0.f;
        for (int off = 1; off < 8; off <<= 1) {
            a  += __shfl_xor(a,  off, 32);
            bb += __shfl_xor(bb, off, 32);
        }
        if (lane == 0) { red[0] = a; red[1] = bb; }
    }
    __syncthreads();
    float mu  = red[0] / H;
    float var = red[1] / H - mu * mu;
    float inv = rsqrtf(var + 1e-5f);
    for (int i = 0; i < 4; ++i) {
        int col = tid * 4 + i;
        float hh = (v[i] - mu) * inv;
        float s  = ss[(size_t)b * 2 * H + col];
        float sh = ss[(size_t)b * 2 * H + H + col];
        outh[(size_t)row * H + col] = (__bf16)((1.f + s) * hh + sh);
    }
}

// ---------------------------------------------------------------------------
// Kernel 4: V transpose: qkv[b,key,2,h,d] -> Vt[b,h,d,key] (bf16).
// One block per (b,h,32-key chunk); LDS bounce for coalesced in/out.
// ---------------------------------------------------------------------------
__global__ __launch_bounds__(256) void vtrans_kernel(const __bf16* __restrict__ qkv,
                                                     __bf16* __restrict__ Vt,
                                                     int B, int L) {
    const int H = NH * HD;
    int blk = blockIdx.x;
    int kc32 = blk % (L / 32);
    int h    = (blk / (L / 32)) % NH;
    int b    = blk / ((L / 32) * NH);
    int k0 = kc32 * 32;
    int tid = threadIdx.x;

    __shared__ __bf16 Vs[32][72];
    {   // coalesced read: [key][d]
        int key = tid >> 3, d = (tid & 7) * 8;
        *(bf16x8*)&Vs[key][d] =
            *(const bf16x8*)&qkv[((size_t)(b * L + k0 + key) * 3 + 2) * H + h * HD + d];
    }
    __syncthreads();
    {   // transposed write: [d][key]
        int d = tid >> 2, kc = (tid & 3) * 8;
        bf16x8 t;
        for (int i = 0; i < 8; ++i) t[i] = Vs[kc + i][d];
        *(bf16x8*)&Vt[(((size_t)b * NH + h) * HD + d) * L + k0 + kc] = t;
    }
}

// ---------------------------------------------------------------------------
// Kernel 5: bf16 WMMA GEMM, double-buffered LDS, TDM-staged A tiles.
// C = epi(A[M,K]bf16 @ W[K,N]f32 + bias); 128x128x32 tiles.
// EPI: 0=bf16, 1=GELU->bf16, 2=+resid->f32
// ---------------------------------------------------------------------------
template <int EPI>
__global__ __launch_bounds__(256) void gemm_bf16(const __bf16* __restrict__ A,
                                                 const float*  __restrict__ W,
                                                 const float*  __restrict__ bias,
                                                 const float*  __restrict__ resid,
                                                 void* __restrict__ Out,
                                                 int M, int N, int K) {
    constexpr int BM = 128, BN = 128, BK = 32;
    constexpr int AP = 40;                       // 80B rows: 16B-aligned, conflict-free
    __shared__ __bf16 As[2][BM][AP];             // [m][k], 2 buffers
    __shared__ __bf16 Bs[2][BN][AP];             // [n][k] transposed weight tiles

    const int bm = blockIdx.y * BM;
    const int bn = blockIdx.x * BN;
    const int tid  = threadIdx.x;
    const int lane = tid & 31;
    const int wave = tid >> 5;
    const int waveM = wave >> 2;
    const int waveN = wave & 3;
    const int half  = lane >> 4;
    const int l15   = lane & 15;

    auto stageA = [&](int k0, int buf) {
#if USE_TDM
        if (wave == 0)
            tdm_load_tile_2d((unsigned)(uintptr_t)&As[buf][0][0],
                             &A[(size_t)bm * K + k0], BK, BM, (unsigned long long)K);
#else
        int r = tid >> 1, c = (tid & 1) * 16;
        const bf16x8* src = (const bf16x8*)&A[(size_t)(bm + r) * K + k0 + c];
        *(bf16x8*)&As[buf][r][c]     = src[0];
        *(bf16x8*)&As[buf][r][c + 8] = src[1];
#endif
    };
    auto stageB = [&](int k0, int buf) {
        int n  = tid >> 1;
        int kh = (tid & 1) * 16;
        const float* src = &W[(size_t)(k0 + kh) * N + bn + n];
        bf16x8 t0, t1;
        for (int i = 0; i < 8; ++i) {
            t0[i] = (__bf16)src[(size_t)i * N];
            t1[i] = (__bf16)src[(size_t)(i + 8) * N];
        }
        *(bf16x8*)&Bs[buf][n][kh]     = t0;
        *(bf16x8*)&Bs[buf][n][kh + 8] = t1;
        if (k0 + BK < K) __builtin_prefetch((const char*)src + (size_t)BK * N * 4, 0, 1);
    };

    v8f acc[4][2] = {};

    stageA(0, 0);
    stageB(0, 0);
#if USE_TDM
    if (wave == 0) __builtin_amdgcn_s_wait_tensorcnt(0);
#endif
    __syncthreads();

    int buf = 0;
    for (int k0 = 0; k0 < K; k0 += BK) {
        int nk = k0 + BK;
        if (nk < K) { stageA(nk, buf ^ 1); stageB(nk, buf ^ 1); }

        bf16x16 af[4];
        for (int mi = 0; mi < 4; ++mi) {
            int row = waveM * 64 + mi * 16 + l15;
            bf16x8 c0 = *(const bf16x8*)&As[buf][row][half * 8];
            bf16x8 c1 = *(const bf16x8*)&As[buf][row][half * 8 + 16];
            af[mi] = __builtin_shufflevector(c0, c1, 0,1,2,3,4,5,6,7,8,9,10,11,12,13,14,15);
        }
        bf16x16 bfr[2];
        for (int ni = 0; ni < 2; ++ni) {
            int col = waveN * 32 + ni * 16 + l15;
            bf16x8 c0 = *(const bf16x8*)&Bs[buf][col][half * 8];
            bf16x8 c1 = *(const bf16x8*)&Bs[buf][col][half * 8 + 16];
            bfr[ni] = __builtin_shufflevector(c0, c1, 0,1,2,3,4,5,6,7,8,9,10,11,12,13,14,15);
        }
        for (int mi = 0; mi < 4; ++mi)
            for (int ni = 0; ni < 2; ++ni)
                acc[mi][ni] = __builtin_amdgcn_wmma_f32_16x16x32_bf16(
                    false, af[mi], false, bfr[ni], (short)0, acc[mi][ni], false, false);

#if USE_TDM
        if (wave == 0 && nk < K) __builtin_amdgcn_s_wait_tensorcnt(0);
#endif
        __syncthreads();
        buf ^= 1;
    }

    for (int mi = 0; mi < 4; ++mi)
        for (int ni = 0; ni < 2; ++ni)
            for (int r = 0; r < 8; ++r) {
                int row = bm + waveM * 64 + mi * 16 + r + half * 8;
                int col = bn + waveN * 32 + ni * 16 + l15;
                float v = acc[mi][ni][r] + bias[col];
                size_t o = (size_t)row * N + col;
                if (EPI == 0) {
                    ((__bf16*)Out)[o] = (__bf16)v;
                } else if (EPI == 1) {
                    float g = 0.5f * v * (1.f + erff(v * 0.70710678f));
                    ((__bf16*)Out)[o] = (__bf16)g;
                } else {
                    ((float*)Out)[o] = v + resid[o];
                }
            }
}

// ---------------------------------------------------------------------------
// Kernel 6: flash attention. 8 waves/block, 16 queries/wave, keys in 32-chunks.
// V comes pre-transposed (Vt[b,h,d,key]) so every fragment is ds_load_b128.
// ---------------------------------------------------------------------------
__global__ __launch_bounds__(256) void attn_kernel(const __bf16* __restrict__ qkv,   // [B,L,3,NH,HD]
                                                   const __bf16* __restrict__ Vt,    // [B,NH,HD,L]
                                                   const __bf16* __restrict__ comb,  // [B,L,L]
                                                   __bf16* __restrict__ out,         // [B,L,H]
                                                   int B, int L) {
    const int H = NH * HD;
    const int qchunks = L / 128;
    int blk = blockIdx.x;
    int qc  = blk % qchunks;
    int h   = (blk / qchunks) % NH;
    int b   = blk / (qchunks * NH);

    const int tid  = threadIdx.x;
    const int lane = tid & 31;
    const int wave = tid >> 5;
    const int half = lane >> 4;
    const int l15  = lane & 15;
    const int qBase = qc * 128 + wave * 16;

    __shared__ __bf16 Qs[128][72];       // [q][d], 144B rows
    __shared__ __bf16 Ks[32][72];        // [key][d]
    __shared__ __bf16 Vts[HD][40];       // [d][key], 80B rows
    __shared__ __bf16 Ps[8][16][40];     // per-wave P re-fragment bounce

    {   // stage Q block-wide (coalesced b128)
        int r = tid >> 1, c = (tid & 1) * 32;
        const bf16x8* src = (const bf16x8*)&qkv[((size_t)(b * L + qc * 128 + r) * 3 + 0) * H + h * HD + c];
        *(bf16x8*)&Qs[r][c]      = src[0];
        *(bf16x8*)&Qs[r][c + 8]  = src[1];
        *(bf16x8*)&Qs[r][c + 16] = src[2];
        *(bf16x8*)&Qs[r][c + 24] = src[3];
    }
    __syncthreads();
    bf16x16 qf[2];
    for (int s = 0; s < 2; ++s) {
        bf16x8 c0 = *(const bf16x8*)&Qs[wave * 16 + l15][s * 32 + half * 8];
        bf16x8 c1 = *(const bf16x8*)&Qs[wave * 16 + l15][s * 32 + half * 8 + 16];
        qf[s] = __builtin_shufflevector(c0, c1, 0,1,2,3,4,5,6,7,8,9,10,11,12,13,14,15);
    }

    v8f o[4] = {};
    float mrow[8], lrow[8];
    for (int r = 0; r < 8; ++r) { mrow[r] = -1e30f; lrow[r] = 0.f; }

    for (int k0 = 0; k0 < L; k0 += 32) {
        {   // stage K [key][d] (one b128/thread) and Vt [d][key] (one b128/thread)
            int key = tid >> 3, d = (tid & 7) * 8;
            *(bf16x8*)&Ks[key][d] =
                *(const bf16x8*)&qkv[((size_t)(b * L + k0 + key) * 3 + 1) * H + h * HD + d];
            int dv = tid >> 2, kc = (tid & 3) * 8;
            *(bf16x8*)&Vts[dv][kc] =
                *(const bf16x8*)&Vt[(((size_t)b * NH + h) * HD + dv) * L + k0 + kc];
        }
        __syncthreads();

        // S = Q K^T : B-layout K fragments, table-order chunks
        v8f sacc[2] = {};
        for (int nt = 0; nt < 2; ++nt)
            for (int ds = 0; ds < 2; ++ds) {
                bf16x8 c0 = *(const bf16x8*)&Ks[nt * 16 + l15][ds * 32 + half * 8];
                bf16x8 c1 = *(const bf16x8*)&Ks[nt * 16 + l15][ds * 32 + half * 8 + 16];
                bf16x16 kf = __builtin_shufflevector(c0, c1, 0,1,2,3,4,5,6,7,8,9,10,11,12,13,14,15);
                sacc[nt] = __builtin_amdgcn_wmma_f32_16x16x32_bf16(
                    false, qf[ds], false, kf, (short)0, sacc[nt], false, false);
            }

        // scale + combined bias
        float sv[2][8];
        for (int nt = 0; nt < 2; ++nt)
            for (int r = 0; r < 8; ++r) {
                int q   = qBase + r + half * 8;
                int key = k0 + nt * 16 + l15;
                sv[nt][r] = sacc[nt][r] * 0.125f
                          + (float)comb[((size_t)b * L + q) * L + key];
            }

        // online softmax (rows live in 16-lane halves)
        float newm[8], rsum[8];
        for (int r = 0; r < 8; ++r) {
            float v = fmaxf(sv[0][r], sv[1][r]);
            for (int off = 1; off < 16; off <<= 1) v = fmaxf(v, __shfl_xor(v, off, 32));
            newm[r] = fmaxf(mrow[r], v);
            rsum[r] = 0.f;
        }
        for (int nt = 0; nt < 2; ++nt)
            for (int r = 0; r < 8; ++r) {
                float p = __expf(sv[nt][r] - newm[r]);
                rsum[r] += p;
                Ps[wave][r + half * 8][nt * 16 + l15] = (__bf16)p;
            }
        for (int r = 0; r < 8; ++r) {
            float v = rsum[r];
            for (int off = 1; off < 16; off <<= 1) v += __shfl_xor(v, off, 32);
            float corr = __expf(mrow[r] - newm[r]);
            lrow[r] = lrow[r] * corr + v;
            mrow[r] = newm[r];
            for (int dt = 0; dt < 4; ++dt) o[dt][r] *= corr;
        }

        // re-fragment P (C-layout -> A-layout)
        bf16x16 pf;
        {
            bf16x8 c0 = *(const bf16x8*)&Ps[wave][l15][half * 8];
            bf16x8 c1 = *(const bf16x8*)&Ps[wave][l15][half * 8 + 16];
            pf = __builtin_shufflevector(c0, c1, 0,1,2,3,4,5,6,7,8,9,10,11,12,13,14,15);
        }

        // O += P @ V : B-fragment from transposed V tile, table-order chunks
        for (int dt = 0; dt < 4; ++dt) {
            bf16x8 c0 = *(const bf16x8*)&Vts[dt * 16 + l15][half * 8];
            bf16x8 c1 = *(const bf16x8*)&Vts[dt * 16 + l15][half * 8 + 16];
            bf16x16 vf = __builtin_shufflevector(c0, c1, 0,1,2,3,4,5,6,7,8,9,10,11,12,13,14,15);
            o[dt] = __builtin_amdgcn_wmma_f32_16x16x32_bf16(
                false, pf, false, vf, (short)0, o[dt], false, false);
        }
        __syncthreads();
    }

    for (int dt = 0; dt < 4; ++dt)
        for (int r = 0; r < 8; ++r) {
            int q = qBase + r + half * 8;
            int d = dt * 16 + l15;
            out[(size_t)(b * L + q) * H + h * HD + d] = (__bf16)(o[dt][r] / lrow[r]);
        }
}

// ---------------------------------------------------------------------------
// Launch
// ---------------------------------------------------------------------------
extern "C" void kernel_launch(void* const* d_in, const int* in_sizes, int n_in,
                              void* d_out, int out_size, void* d_ws, size_t ws_size,
                              hipStream_t stream) {
    const float* x         = (const float*)d_in[0];
    const float* attn_mask = (const float*)d_in[1];
    const float* attn_bias = (const float*)d_in[2];
    const float* t_vec     = (const float*)d_in[3];
    const unsigned char* kpm = (const unsigned char*)d_in[4];
    const float* Wqkv  = (const float*)d_in[5];
    const float* bqkv  = (const float*)d_in[6];
    const float* Wproj = (const float*)d_in[7];
    const float* bproj = (const float*)d_in[8];
    const float* Wss1  = (const float*)d_in[9];
    const float* bss1  = (const float*)d_in[10];
    const float* Wss2  = (const float*)d_in[11];
    const float* bss2  = (const float*)d_in[12];
    const float* Wmlp1 = (const float*)d_in[13];
    const float* bmlp1 = (const float*)d_in[14];
    const float* Wmlp2 = (const float*)d_in[15];
    const float* bmlp2 = (const float*)d_in[16];
    float* out = (float*)d_out;

    const int B = 2, L = 2048, H = 1024, Dff = 4096;
    const int M = B * L;

    char* p = (char*)d_ws;
    float*  ss1    = (float*)p;  p += (size_t)B * 2 * H * 4;
    float*  ss2    = (float*)p;  p += (size_t)B * 2 * H * 4;
    __bf16* h1     = (__bf16*)p; p += (size_t)M * H * 2;
    __bf16* qkvb   = (__bf16*)p; p += (size_t)M * 3 * H * 2;
    __bf16* attn_o = (__bf16*)p; p += (size_t)M * H * 2;
    float*  x1     = (float*)p;  p += (size_t)M * H * 4;
    __bf16* h2     = (__bf16*)p; p += (size_t)M * H * 2;
    __bf16* m1     = (__bf16*)p; p += (size_t)M * Dff * 2;
    __bf16* comb   = (__bf16*)p; p += (size_t)B * L * L * 2;
    __bf16* Vt     = (__bf16*)p; p += (size_t)M * H * 2;

    cond_kernel<<<(B * 2 * H + 255) / 256, 256, 0, stream>>>(
        t_vec, Wss1, bss1, Wss2, bss2, ss1, ss2, B, H);

    biascomb_kernel<<<(int)(((size_t)B * L * L) / 256), 256, 0, stream>>>(
        attn_mask, attn_bias, kpm, comb, B, L);

    adaln_kernel<<<M, 256, 0, stream>>>(x, ss1, h1, L, H);

    {   // QKV projection
        dim3 g(3 * H / 128, M / 128);
        gemm_bf16<0><<<g, 256, 0, stream>>>(h1, Wqkv, bqkv, nullptr, qkvb, M, 3 * H, H);
    }

    vtrans_kernel<<<B * NH * (L / 32), 256, 0, stream>>>(qkvb, Vt, B, L);

    attn_kernel<<<B * NH * (L / 128), 256, 0, stream>>>(qkvb, Vt, comb, attn_o, B, L);

    {   // output projection + residual
        dim3 g(H / 128, M / 128);
        gemm_bf16<2><<<g, 256, 0, stream>>>(attn_o, Wproj, bproj, x, x1, M, H, H);
    }

    adaln_kernel<<<M, 256, 0, stream>>>(x1, ss2, h2, L, H);

    {   // MLP up + GELU
        dim3 g(Dff / 128, M / 128);
        gemm_bf16<1><<<g, 256, 0, stream>>>(h2, Wmlp1, bmlp1, nullptr, m1, M, Dff, H);
    }

    {   // MLP down + residual -> out
        dim3 g(H / 128, M / 128);
        gemm_bf16<2><<<g, 256, 0, stream>>>(m1, Wmlp2, bmlp2, x1, out, M, H, Dff);
    }
}
